// ChunkedDiagonalMLP_16569983828097
// MI455X (gfx1250) — matrix operand order repaired and checked by reading, not compiled, not run on hardware
//
#include <hip/hip_runtime.h>
#include <hip/hip_bf16.h>
#include <math.h>

#define D_MODEL 4096
#define N_CHUNKS 8
#define CHUNK 512
#define NTOK 16384            // B*L = 4*4096
#define M_TILE 32             // tokens per block
#define LDP 520               // LDS row pitch (bf16 elems); 260 dwords -> conflict-free b128 reads

typedef __bf16 bf16;
typedef __attribute__((ext_vector_type(16))) __bf16 v16bf;
typedef __attribute__((ext_vector_type(8)))  __bf16 v8bf;
typedef __attribute__((ext_vector_type(4)))  __bf16 v4bf;
typedef __attribute__((ext_vector_type(8)))  float  v8f;

// A-fragment (16x32 bf16) from row-major LDS tile.
// lanes 0-15 hold M=lane, K in {0..15}; lanes 16-31 hold K in {16..31}.
__device__ __forceinline__ v16bf load_a_frag(const bf16* sm, int rbase, int kbase, int lane) {
  int lrow  = lane & 15;
  int khalf = lane >> 4;                       // 0 or 1
  const bf16* p = sm + (rbase + lrow) * LDP + kbase + khalf * 16;
  v8bf a0 = *(const v8bf*)p;                   // K = kbase + khalf*16 .. +7
  v8bf a1 = *(const v8bf*)(p + 8);             // K = kbase + khalf*16 + 8 .. +15
  return __builtin_shufflevector(a0, a1, 0,1,2,3,4,5,6,7,8,9,10,11,12,13,14,15);
}

// B-fragment (32x16 bf16) from transposed weights [d][k] (K contiguous).
// lanes 0-15: N=lane, K=kbase..kbase+15 ; lanes 16-31: N=lane-16, K=kbase+16..kbase+31.
__device__ __forceinline__ v16bf load_b_frag(const bf16* wt, int dbase, int kbase, int lane) {
  int ncol = lane & 15;
  int kh   = (lane >> 4) << 4;                 // 0 or 16
  return *(const v16bf*)(wt + (size_t)(dbase + ncol) * CHUNK + kbase + kh);
}

// fp32 [n][c][d] -> bf16 transposed [n][d][c] for both weight tensors.
__global__ __launch_bounds__(256) void prep_w(const float* __restrict__ w1,
                                              const float* __restrict__ w2,
                                              bf16* __restrict__ w1t,
                                              bf16* __restrict__ w2t) {
  size_t idx = (size_t)blockIdx.x * 256 + threadIdx.x;  // dest index: [n][d][c]
  int n = (int)(idx >> 18);
  int d = (int)((idx >> 9) & 511);
  int c = (int)(idx & 511);
  size_t src = ((size_t)n << 18) + ((size_t)c << 9) + (size_t)d;
  w1t[idx] = (bf16)w1[src];
  w2t[idx] = (bf16)w2[src];
}

// Fused per-chunk MLP: out = x + (gelu(x_c @ W1 + b1) @ W2 + b2)
__global__ __launch_bounds__(256) void mlp_kernel(const float* __restrict__ x,
                                                  const bf16*  __restrict__ w1t,
                                                  const float* __restrict__ b1,
                                                  const bf16*  __restrict__ w2t,
                                                  const float* __restrict__ b2,
                                                  float* __restrict__ out) {
  __shared__ bf16 sm[M_TILE * LDP];            // 33,280 B; reused for X then H

  const int chunk = blockIdx.y;
  const int tok0  = blockIdx.x * M_TILE;
  const int tid   = threadIdx.x;
  const int lane  = tid & 31;
  const int wv    = tid >> 5;                  // 8 waves
  const int rtile = (wv & 1) * 16;             // local row base (two 16-row tiles)
  const int cbase = (wv >> 1) * 128;           // local col base (four 128-col slabs)

  // ---- Phase 0: stage X tile (fp32 -> bf16) into LDS, row-major [32][512] ----
  {
    const float* xs = x + (size_t)tok0 * D_MODEL + (size_t)chunk * CHUNK;
    #pragma unroll
    for (int i = 0; i < (M_TILE * CHUNK) / (256 * 4); ++i) {
      int linear = (i * 256 + tid) * 4;
      int row = linear >> 9;
      int col = linear & 511;
      float4 v = *(const float4*)(xs + (size_t)row * D_MODEL + col);
      v4bf bv; bv[0] = (bf16)v.x; bv[1] = (bf16)v.y; bv[2] = (bf16)v.z; bv[3] = (bf16)v.w;
      *(v4bf*)(sm + row * LDP + col) = bv;
    }
  }
  __syncthreads();

  const bf16* w1p = w1t + (size_t)chunk * CHUNK * CHUNK;
  const bf16* w2p = w2t + (size_t)chunk * CHUNK * CHUNK;

  bf16 hreg[8][8];                             // GELU(H) staged in registers

  // ---- Phase 1: H = gelu(X @ W1 + b1) ----
  // k outer / ct inner: each A-fragment loaded from LDS once, reused by 8 WMMAs.
  {
    v8f acc[8] = {};
    #pragma unroll 2
    for (int k = 0; k < 16; ++k) {
      v16bf a = load_a_frag(sm, rtile, k * 32, lane);
      #pragma unroll
      for (int ct = 0; ct < 8; ++ct) {
        v16bf b = load_b_frag(w1p, cbase + ct * 16, k * 32, lane);
        acc[ct] = __builtin_amdgcn_wmma_f32_16x16x32_bf16(false, a, false, b, (short)0,
                                                          acc[ct], false, false);
      }
    }
    #pragma unroll
    for (int ct = 0; ct < 8; ++ct) {
      float bias = b1[chunk * CHUNK + cbase + ct * 16 + (lane & 15)];
      #pragma unroll
      for (int j = 0; j < 8; ++j) {
        float v = acc[ct][j] + bias;
        float g = 0.5f * v * (1.0f + erff(v * 0.70710678118654752f));  // exact-erf gelu
        hreg[ct][j] = (bf16)g;
      }
    }
  }
  __syncthreads();                             // everyone done reading X

  // ---- Store H into the reused LDS buffer (C/D layout -> row-major) ----
  {
    const int scol = lane & 15;
    const int roff = rtile + ((lane >> 4) << 3);   // +8 rows for upper half-wave
    #pragma unroll
    for (int ct = 0; ct < 8; ++ct)
      #pragma unroll
      for (int j = 0; j < 8; ++j)
        sm[(roff + j) * LDP + cbase + ct * 16 + scol] = hreg[ct][j];
  }
  __syncthreads();

  // ---- Phase 2: OUT = H @ W2 + b2 + x (residual), fp32 store ----
  {
    v8f acc[8] = {};
    #pragma unroll 2
    for (int k = 0; k < 16; ++k) {
      v16bf a = load_a_frag(sm, rtile, k * 32, lane);
      #pragma unroll
      for (int ct = 0; ct < 8; ++ct) {
        v16bf b = load_b_frag(w2p, cbase + ct * 16, k * 32, lane);
        acc[ct] = __builtin_amdgcn_wmma_f32_16x16x32_bf16(false, a, false, b, (short)0,
                                                          acc[ct], false, false);
      }
    }
    const int rowb = rtile + ((lane >> 4) << 3);
    #pragma unroll
    for (int ct = 0; ct < 8; ++ct) {
      const int gcol = chunk * CHUNK + cbase + ct * 16 + (lane & 15);
      const float bias = b2[gcol];
      #pragma unroll
      for (int j = 0; j < 8; ++j) {
        size_t idx = (size_t)(tok0 + rowb + j) * D_MODEL + gcol;
        out[idx] = acc[ct][j] + bias + x[idx];
      }
    }
  }
}

// In-place LayerNorm over d=4096, one block per token (float4 vectorized).
__global__ __launch_bounds__(256) void ln_kernel(float* __restrict__ out,
                                                 const float* __restrict__ g,
                                                 const float* __restrict__ b) {
  __shared__ float rsum[256];
  __shared__ float rsum2[256];
  const int tid = threadIdx.x;
  float* row = out + (size_t)blockIdx.x * D_MODEL;
  float4 vals[4];
  float s = 0.f, s2 = 0.f;
  #pragma unroll
  for (int i = 0; i < 4; ++i) {
    float4 v = *(const float4*)(row + (tid + i * 256) * 4);
    vals[i] = v;
    s  += v.x + v.y + v.z + v.w;
    s2 += v.x * v.x + v.y * v.y + v.z * v.z + v.w * v.w;
  }
  rsum[tid] = s; rsum2[tid] = s2;
  __syncthreads();
  for (int off = 128; off > 0; off >>= 1) {
    if (tid < off) { rsum[tid] += rsum[tid + off]; rsum2[tid] += rsum2[tid + off]; }
    __syncthreads();
  }
  const float mu   = rsum[0] * (1.0f / D_MODEL);
  const float var  = rsum2[0] * (1.0f / D_MODEL) - mu * mu;
  const float rinv = rsqrtf(var + 1e-5f);
  #pragma unroll
  for (int i = 0; i < 4; ++i) {
    int d = (tid + i * 256) * 4;
    float4 gv = *(const float4*)(g + d);
    float4 bv = *(const float4*)(b + d);
    float4 v  = vals[i];
    float4 o;
    o.x = (v.x - mu) * rinv * gv.x + bv.x;
    o.y = (v.y - mu) * rinv * gv.y + bv.y;
    o.z = (v.z - mu) * rinv * gv.z + bv.z;
    o.w = (v.w - mu) * rinv * gv.w + bv.w;
    *(float4*)(row + d) = o;
  }
}

extern "C" void kernel_launch(void* const* d_in, const int* in_sizes, int n_in,
                              void* d_out, int out_size, void* d_ws, size_t ws_size,
                              hipStream_t stream) {
  const float* x    = (const float*)d_in[0];
  const float* w1   = (const float*)d_in[1];
  const float* b1   = (const float*)d_in[2];
  const float* w2   = (const float*)d_in[3];
  const float* b2   = (const float*)d_in[4];
  const float* ln_g = (const float*)d_in[5];
  const float* ln_b = (const float*)d_in[6];
  float* out = (float*)d_out;

  // Workspace: bf16 transposed weights, 2 x 8*512*512*2 B = 8 MB total.
  bf16* w1t = (bf16*)d_ws;
  bf16* w2t = w1t + (size_t)N_CHUNKS * CHUNK * CHUNK;

  prep_w<<<(N_CHUNKS * CHUNK * CHUNK) / 256, 256, 0, stream>>>(w1, w2, w1t, w2t);

  dim3 grid(NTOK / M_TILE, N_CHUNKS);
  mlp_kernel<<<grid, 256, 0, stream>>>(x, w1t, b1, w2t, b2, out);

  ln_kernel<<<NTOK, 256, 0, stream>>>(out, ln_g, ln_b);
}
